// HATAFormerMultiHeadAttention_10496900071840
// MI455X (gfx1250) — compile-verified, matrix operand order checked by reading, not run on hardware
//
#include <hip/hip_runtime.h>
#include <hip/hip_bf16.h>
#include <math.h>

// ---------------- problem constants (match reference) ----------------
#define B_   2
#define S_   2048
#define D_   1024
#define H_   16
#define DK_  64
#define WIN_ 64
#define EPS_ 1e-5f

typedef __attribute__((ext_vector_type(16))) _Float16 v16h;
typedef __attribute__((ext_vector_type(8)))  _Float16 v8h;
typedef __attribute__((ext_vector_type(8)))  float    v8f;
typedef __attribute__((ext_vector_type(4)))  float    v4f;

__device__ __forceinline__ v8f zero_v8f() {
  v8f z;
#pragma unroll
  for (int i = 0; i < 8; ++i) z[i] = 0.0f;
  return z;
}

// 16x32 f16 WMMA fragment (A layout; B layout identical with lane = column)
// from a row-major f16 matrix. `p` = base + (lane&15)*ld + kb,
// kb = ((lane>>4)&1)*8.  elem j<8 -> K=kb+j ; elem j>=8 -> K=16+kb+(j-8).
// Two 16-byte loads, zero converts.
__device__ __forceinline__ v16h load_frag_pair_h(const _Float16* __restrict__ p) {
  v8h a0 = *(const v8h*)(p);
  v8h a1 = *(const v8h*)(p + 16);
  v16h h;
#pragma unroll
  for (int j = 0; j < 8; ++j) { h[j] = a0[j]; h[8 + j] = a1[j]; }
  return h;
}

#define WMMA_F32_F16(a, b, c) \
  __builtin_amdgcn_wmma_f32_16x16x32_f16(false, (a), false, (b), (short)0, (c), false, false)

// =====================================================================
// Kernel 0: streaming f32 -> f16 convert (hoists all GEMM converts out of
// the O(M*N*K) loops; done once per operand). n must be a multiple of 2048.
// =====================================================================
__global__ __launch_bounds__(256) void f32_to_f16_kernel(
    const float* __restrict__ src, _Float16* __restrict__ dst) {
  const int i = (blockIdx.x * 256 + threadIdx.x) * 8;
  v4f x0 = *(const v4f*)(src + i);
  v4f x1 = *(const v4f*)(src + i + 4);
  v8h o;
#pragma unroll
  for (int j = 0; j < 4; ++j) {
    o[j]     = (_Float16)x0[j];
    o[4 + j] = (_Float16)x1[j];
  }
  *(v8h*)(dst + i) = o;
}

// =====================================================================
// Kernel 1/4: Y = X[M,K] @ W[N,K]^T + bias  (nn.Linear), f16 operands.
// Inner loop is pure b128-load + v_wmma. Optional f32 / f16 outputs.
// Block = 256 thr = 8 waves; wave -> 16 rows x 64 cols; grid (N/64, M/128).
// =====================================================================
__global__ __launch_bounds__(256) void gemm_h_xwT_kernel(
    const _Float16* __restrict__ X, const _Float16* __restrict__ W,
    const float* __restrict__ bias, float* __restrict__ Yf,
    _Float16* __restrict__ Yh, int M, int N, int K) {
  const int lane = threadIdx.x & 31;
  const int wid  = threadIdx.x >> 5;
  const int m0 = (blockIdx.y * 8 + wid) * 16;
  const int n0 = blockIdx.x * 64;

  const int arow = lane & 15;
  const int kb   = ((lane >> 4) & 1) << 3;
  const _Float16* Xrow = X + (size_t)(m0 + arow) * K + kb;
  const _Float16* Wrow = W + (size_t)(n0 + arow) * K + kb;

  v8f acc[4];
#pragma unroll
  for (int dt = 0; dt < 4; ++dt) acc[dt] = zero_v8f();

  for (int k0 = 0; k0 < K; k0 += 32) {
    __builtin_prefetch(Xrow + k0 + 128, 0, 1);
    v16h a = load_frag_pair_h(Xrow + k0);
#pragma unroll
    for (int dt = 0; dt < 4; ++dt) {
      v16h bf = load_frag_pair_h(Wrow + (size_t)dt * 16 * K + k0);
      acc[dt] = WMMA_F32_F16(a, bf, acc[dt]);
    }
  }

  const int mhi = ((lane >> 4) & 1) << 3;
#pragma unroll
  for (int dt = 0; dt < 4; ++dt) {
    const int col = n0 + dt * 16 + (lane & 15);
    const float bv = bias[col];
#pragma unroll
    for (int r = 0; r < 8; ++r) {
      const int row = m0 + r + mhi;
      const float val = acc[dt][r] + bv;
      if (Yf) Yf[(size_t)row * N + col] = val;
      if (Yh) Yh[(size_t)row * N + col] = (_Float16)val;
    }
  }
}

// =====================================================================
// Kernel 2: raw biased scores -> attn region of d_out (f16 Q/K inputs).
// scores[q,k] = (Q.K)/8 + (|i-j|<=WIN) * sigmoid(local_bias[h])
// grid = (S/128, B*H); wave -> one 16-row q tile, loops all k tiles.
// =====================================================================
__global__ __launch_bounds__(256) void attn_scores_kernel(
    const _Float16* __restrict__ qh, const _Float16* __restrict__ kh,
    const float* __restrict__ local_bias, float* __restrict__ attn) {
  const int lane = threadIdx.x & 31;
  const int wid  = threadIdx.x >> 5;
  const int bh = blockIdx.y;
  const int b = bh >> 4, h = bh & 15;
  const int q0 = (blockIdx.x * 8 + wid) * 16;
  const float sig = 1.0f / (1.0f + __expf(-local_bias[h]));
  const float scale = 0.125f;  // 1/sqrt(64)

  const int arow = lane & 15;
  const int kb   = ((lane >> 4) & 1) << 3;
  const int mhi  = ((lane >> 4) & 1) << 3;

  const _Float16* Qrow = qh + ((size_t)(b * S_ + q0 + arow) * D_ + h * DK_) + kb;
  v16h a0 = load_frag_pair_h(Qrow);
  v16h a1 = load_frag_pair_h(Qrow + 32);

  float* attn_bh = attn + (size_t)bh * S_ * S_;

  for (int k0 = 0; k0 < S_; k0 += 16) {
    const _Float16* Krow = kh + ((size_t)(b * S_ + k0 + arow) * D_ + h * DK_) + kb;
    __builtin_prefetch(Krow + 16 * D_, 0, 1);
    v16h b0 = load_frag_pair_h(Krow);
    v16h b1 = load_frag_pair_h(Krow + 32);
    v8f c = zero_v8f();
    c = WMMA_F32_F16(a0, b0, c);
    c = WMMA_F32_F16(a1, b1, c);

    const int col = k0 + (lane & 15);
#pragma unroll
    for (int r = 0; r < 8; ++r) {
      const int row = q0 + r + mhi;
      int dd = row - col; if (dd < 0) dd = -dd;
      const float t = c[r] * scale + ((dd <= WIN_) ? sig : 0.0f);
      attn_bh[(size_t)row * S_ + col] = t;
    }
  }
}

// =====================================================================
// Kernel 3: per-row softmax stats (max, sum exp). One wave per row.
// =====================================================================
__global__ __launch_bounds__(256) void softmax_stats_kernel(
    const float* __restrict__ attn, float* __restrict__ stats) {
  const int lane = threadIdx.x & 31;
  const int wid  = threadIdx.x >> 5;
  const size_t row = (size_t)blockIdx.x * 8 + wid;
  const float* rp = attn + row * S_;

  float m = -1e30f;
  for (int t = lane; t < S_; t += 32) m = fmaxf(m, rp[t]);
#pragma unroll
  for (int off = 16; off > 0; off >>= 1) m = fmaxf(m, __shfl_xor(m, off, 32));

  float l = 0.0f;
  for (int t = lane; t < S_; t += 32) l += __expf(rp[t] - m);
#pragma unroll
  for (int off = 16; off > 0; off >>= 1) l += __shfl_xor(l, off, 32);

  if (lane == 0) { stats[row * 2] = m; stats[row * 2 + 1] = l; }
}

// =====================================================================
// Kernel 4: finalize probabilities (write back to d_out) and ctx = P @ V.
// V tile (32 k x 64 d, f16) staged into LDS with the CDNA5 async engine:
// one global_load_async_to_lds_b128 per thread (4 KB tile), completed with
// s_wait_asynccnt + barrier, shared by all 8 waves. ctx emitted as f16 so
// the output-projection GEMM needs no convert pass.
// grid = (S/128, B*H), block = 256.
// =====================================================================
__global__ __launch_bounds__(256) void attn_ctx_kernel(
    const _Float16* __restrict__ vh, const float* __restrict__ stats,
    float* __restrict__ attn, _Float16* __restrict__ ctxh) {
  __shared__ _Float16 vtile[32 * 64];  // [k][d] row-major, 4 KB

  const int lane = threadIdx.x & 31;
  const int wid  = threadIdx.x >> 5;
  const int bh = blockIdx.y;
  const int b = bh >> 4, h = bh & 15;
  const int q0 = (blockIdx.x * 8 + wid) * 16;
  const int arow = lane & 15;
  const int kb   = ((lane >> 4) & 1) << 3;
  const int mhi  = ((lane >> 4) & 1) << 3;

  // per-thread async-copy source mapping: 16 bytes each, tile-contiguous LDS
  const unsigned ti = threadIdx.x * 16u;       // byte index into 4 KB tile
  const int      tk = (int)(ti >> 7);          // row k (128 B per row)
  const unsigned trb = ti & 127u;              // byte offset within row
  const unsigned lds_a = (unsigned)(uintptr_t)(&vtile[0]) + ti;

  float* attn_row = attn + ((size_t)bh * S_ + q0 + arow) * S_;
  const float m    = stats[((size_t)bh * S_ + q0 + arow) * 2 + 0];
  const float invl = 1.0f / stats[((size_t)bh * S_ + q0 + arow) * 2 + 1];

  v8f acc[4];
#pragma unroll
  for (int dt = 0; dt < 4; ++dt) acc[dt] = zero_v8f();

  for (int k0 = 0; k0 < S_; k0 += 32) {
    __syncthreads();  // protect previous iteration's LDS reads
    {
      const _Float16* src =
          vh + ((size_t)(b * S_ + k0 + tk) * D_ + h * DK_) + (trb >> 1);
      const unsigned long long ga = (unsigned long long)(uintptr_t)src;
      asm volatile("global_load_async_to_lds_b128 %0, %1, off"
                   :: "v"(lds_a), "v"(ga) : "memory");
    }
    asm volatile("s_wait_asynccnt 0x0" ::: "memory");
    __syncthreads();

    // A fragment: p = exp(s - m) / l ; final probabilities written back out.
    float* sp = attn_row + k0 + kb;
    v4f x0 = *(const v4f*)(sp + 0);
    v4f x1 = *(const v4f*)(sp + 4);
    v4f x2 = *(const v4f*)(sp + 16);
    v4f x3 = *(const v4f*)(sp + 20);
    v16h a;
#pragma unroll
    for (int j = 0; j < 4; ++j) {
      x0[j] = __expf(x0[j] - m) * invl;
      x1[j] = __expf(x1[j] - m) * invl;
      x2[j] = __expf(x2[j] - m) * invl;
      x3[j] = __expf(x3[j] - m) * invl;
      a[j]      = (_Float16)x0[j];
      a[4 + j]  = (_Float16)x1[j];
      a[8 + j]  = (_Float16)x2[j];
      a[12 + j] = (_Float16)x3[j];
    }
    *(v4f*)(sp + 0)  = x0;
    *(v4f*)(sp + 4)  = x1;
    *(v4f*)(sp + 16) = x2;
    *(v4f*)(sp + 20) = x3;

    // B fragments straight from the async-staged f16 tile (lane = column d).
#pragma unroll
    for (int dt = 0; dt < 4; ++dt) {
      const int d = dt * 16 + arow;
      v16h bf;
#pragma unroll
      for (int j = 0; j < 8; ++j) {
        bf[j]     = vtile[(kb + j) * 64 + d];
        bf[8 + j] = vtile[(16 + kb + j) * 64 + d];
      }
      acc[dt] = WMMA_F32_F16(a, bf, acc[dt]);
    }
  }

#pragma unroll
  for (int dt = 0; dt < 4; ++dt) {
#pragma unroll
    for (int r = 0; r < 8; ++r) {
      const int row = q0 + r + mhi;
      ctxh[(size_t)(b * S_ + row) * D_ + h * DK_ + dt * 16 + (lane & 15)] =
          (_Float16)acc[dt][r];
    }
  }
}

// =====================================================================
// Kernel 5: y = LayerNorm(oproj + qp) * gamma + beta. One block per row.
// =====================================================================
__global__ __launch_bounds__(256) void layernorm_kernel(
    const float* __restrict__ oproj, const float* __restrict__ qp,
    const float* __restrict__ gamma, const float* __restrict__ beta,
    float* __restrict__ y) {
  __shared__ float red[256];
  const int row = blockIdx.x;
  const int tid = threadIdx.x;
  const float* o = oproj + (size_t)row * D_;
  const float* q = qp + (size_t)row * D_;

  float x[4];
  float s = 0.0f, ss = 0.0f;
#pragma unroll
  for (int j = 0; j < 4; ++j) {
    const int idx = tid + j * 256;
    x[j] = o[idx] + q[idx];
    s += x[j];
    ss += x[j] * x[j];
  }

  red[tid] = s; __syncthreads();
  for (int off = 128; off > 0; off >>= 1) {
    if (tid < off) red[tid] += red[tid + off];
    __syncthreads();
  }
  const float mu = red[0] * (1.0f / D_);
  __syncthreads();

  red[tid] = ss; __syncthreads();
  for (int off = 128; off > 0; off >>= 1) {
    if (tid < off) red[tid] += red[tid + off];
    __syncthreads();
  }
  const float var = red[0] * (1.0f / D_) - mu * mu;
  const float rinv = rsqrtf(var + EPS_);

#pragma unroll
  for (int j = 0; j < 4; ++j) {
    const int idx = tid + j * 256;
    y[(size_t)row * D_ + idx] = (x[j] - mu) * rinv * gamma[idx] + beta[idx];
  }
}

// =====================================================================
// d_out = [ y : B*S*D f32 | attn : B*H*S*S f32 ]
// workspace (stream-ordered reuse of wh/xh across GEMMs):
//   qp f32 | oproj f32 | stats | qh,kh,vh f16 | wh f16 (1 weight) | xh f16
//   = 16 + 16 + 0.5 + 24 + 2 + 8  ~= 66.5 MB
// =====================================================================
extern "C" void kernel_launch(void* const* d_in, const int* in_sizes, int n_in,
                              void* d_out, int out_size, void* d_ws, size_t ws_size,
                              hipStream_t stream) {
  (void)in_sizes; (void)n_in; (void)out_size; (void)ws_size;
  const float* q_in = (const float*)d_in[0];
  const float* k_in = (const float*)d_in[1];
  const float* v_in = (const float*)d_in[2];
  const float* Wq = (const float*)d_in[3];
  const float* bq = (const float*)d_in[4];
  const float* Wk = (const float*)d_in[5];
  const float* bk = (const float*)d_in[6];
  const float* Wv = (const float*)d_in[7];
  const float* bv = (const float*)d_in[8];
  const float* Wo = (const float*)d_in[9];
  const float* bo = (const float*)d_in[10];
  const float* local_bias = (const float*)d_in[11];
  const float* gamma = (const float*)d_in[12];
  const float* beta  = (const float*)d_in[13];

  const size_t MSD = (size_t)B_ * S_ * D_;  // 4,194,304 elements
  const size_t DD  = (size_t)D_ * D_;       // 1,048,576 elements
  char* w = (char*)d_ws;
  float* qp    = (float*)w;      w += MSD * sizeof(float);
  float* oproj = (float*)w;      w += MSD * sizeof(float);
  float* stats = (float*)w;      w += (size_t)B_ * H_ * S_ * 2 * sizeof(float);
  _Float16* qh = (_Float16*)w;   w += MSD * sizeof(_Float16);
  _Float16* kh = (_Float16*)w;   w += MSD * sizeof(_Float16);
  _Float16* vh = (_Float16*)w;   w += MSD * sizeof(_Float16);
  _Float16* wh = (_Float16*)w;   w += DD * sizeof(_Float16);   // reused 4x
  _Float16* xh = (_Float16*)w;                                  // reused 4x

  float* y    = (float*)d_out;
  float* attn = y + MSD;

  const int M = B_ * S_;                       // 4096
  const dim3 gemm_grid(D_ / 64, M / 128);      // (16, 32)
  const dim3 attn_grid(S_ / 128, B_ * H_);     // (16, 32)
  const int cvt_x = (int)(MSD / 2048);         // activations
  const int cvt_w = (int)(DD / 2048);          // weights

  // ---- Q projection ----
  f32_to_f16_kernel<<<cvt_x, 256, 0, stream>>>(q_in, xh);
  f32_to_f16_kernel<<<cvt_w, 256, 0, stream>>>(Wq, wh);
  gemm_h_xwT_kernel<<<gemm_grid, 256, 0, stream>>>(xh, wh, bq, qp, qh, M, D_, D_);
  // ---- K projection ----
  f32_to_f16_kernel<<<cvt_x, 256, 0, stream>>>(k_in, xh);
  f32_to_f16_kernel<<<cvt_w, 256, 0, stream>>>(Wk, wh);
  gemm_h_xwT_kernel<<<gemm_grid, 256, 0, stream>>>(xh, wh, bk, (float*)nullptr, kh, M, D_, D_);
  // ---- V projection ----
  f32_to_f16_kernel<<<cvt_x, 256, 0, stream>>>(v_in, xh);
  f32_to_f16_kernel<<<cvt_w, 256, 0, stream>>>(Wv, wh);
  gemm_h_xwT_kernel<<<gemm_grid, 256, 0, stream>>>(xh, wh, bv, (float*)nullptr, vh, M, D_, D_);

  // ---- attention ----
  attn_scores_kernel<<<attn_grid, 256, 0, stream>>>(qh, kh, local_bias, attn);
  softmax_stats_kernel<<<(B_ * H_ * S_) / 8, 256, 0, stream>>>(attn, stats);
  attn_ctx_kernel<<<attn_grid, 256, 0, stream>>>(vh, stats, attn, xh /*ctx f16*/);

  // ---- output projection + LayerNorm ----
  f32_to_f16_kernel<<<cvt_w, 256, 0, stream>>>(Wo, wh);
  gemm_h_xwT_kernel<<<gemm_grid, 256, 0, stream>>>(xh, wh, bo, oproj, (_Float16*)nullptr, M, D_, D_);
  layernorm_kernel<<<M, 256, 0, stream>>>(oproj, qp, gamma, beta, y);
}
